// Attention_73108933312525
// MI455X (gfx1250) — matrix operand order repaired and checked by reading, not compiled, not run on hardware
//
#include <hip/hip_runtime.h>

#define HEADS  8
#define DH     64
#define DMODEL 512
#define SEG_LMAX 1536

typedef __attribute__((ext_vector_type(16))) __bf16 bf16x16;
typedef __attribute__((ext_vector_type(8)))  float  f32x8;

union FragAB {
    uint4   q[2];
    bf16x16 v;
};

__device__ __forceinline__ unsigned short f2bf(float f) {
    unsigned int u = __float_as_uint(f);
    u += 0x7FFFu + ((u >> 16) & 1u);   // round-to-nearest-even
    return (unsigned short)(u >> 16);
}

// Async 16-byte global->LDS copy (CDNA5 VGLOBAL async op, tracked by ASYNCcnt).
// lds generic pointer: low 32 bits are the wave-relative LDS byte offset.
__device__ __forceinline__ void async_copy_b128(void* lds, const void* gaddr) {
    unsigned lds_off = (unsigned)(size_t)lds;
    asm volatile("global_load_async_to_lds_b128 %0, %1, off"
                 :: "v"(lds_off), "v"(gaddr) : "memory");
}

__device__ __forceinline__ void wait_asynccnt0() {
#if __has_builtin(__builtin_amdgcn_s_wait_asynccnt)
    __builtin_amdgcn_s_wait_asynccnt(0);
#else
    asm volatile("s_wait_asynccnt 0" ::: "memory");
#endif
}

// ---------------------------------------------------------------------------
// C[M,512] = A[M,512] * W[512,512]; block tile 128x64, 8 waves, wave tile 16x64.
// K staged 64 at a time -> 8 WMMAs per barrier pair per wave.
// mode 0: store bf16 (ushort) to out. mode 1: store f32 + bias to out.
// ---------------------------------------------------------------------------
__global__ void __launch_bounds__(256)
gemm512_wmma(const float* __restrict__ A, const float* __restrict__ W,
             const float* __restrict__ bias, void* __restrict__ out, int mode)
{
    __shared__ alignas(16) unsigned short Abuf[128 * 64]; // [row][k] bf16 (16KB)
    __shared__ alignas(16) unsigned short Btb[64 * 64];   // [n][k]   bf16 (8KB)

    const int tid  = threadIdx.x;
    const int lane = tid & 31;
    const int wave = tid >> 5;
    const int half = lane >> 4;
    const int ln   = lane & 15;
    const int bm   = blockIdx.x * 128;
    const int bn   = blockIdx.y * 64;

    f32x8 acc[4];
#pragma unroll
    for (int j = 0; j < 4; ++j)
#pragma unroll
        for (int r = 0; r < 8; ++r) acc[j][r] = 0.0f;

    const int arow = wave * 16 + ln;

    for (int kk = 0; kk < DMODEL; kk += 64) {
        // stage A tile 128x64 f32 -> bf16 LDS (2048 float4 / 256 thr)
#pragma unroll
        for (int i = 0; i < 8; ++i) {
            int idx = tid + i * 256;
            int row = idx >> 4;
            int c4  = idx & 15;
            float4 f = *(const float4*)&A[(size_t)(bm + row) * DMODEL + kk + c4 * 4];
            unsigned short* p = &Abuf[row * 64 + c4 * 4];
            p[0] = f2bf(f.x); p[1] = f2bf(f.y); p[2] = f2bf(f.z); p[3] = f2bf(f.w);
        }
        // stage W tile 64x64 transposed -> Btb[n][k]
#pragma unroll
        for (int i = 0; i < 16; ++i) {
            int idx = tid + i * 256;
            int k = idx >> 6;
            int n = idx & 63;
            Btb[n * 64 + k] = f2bf(W[(size_t)(kk + k) * DMODEL + bn + n]);
        }
        __syncthreads();

#pragma unroll
        for (int c = 0; c < 2; ++c) {
            FragAB a;
            a.q[0] = *(const uint4*)&Abuf[arow * 64 + c * 32 + 8 * half];
            a.q[1] = *(const uint4*)&Abuf[arow * 64 + c * 32 + 16 + 8 * half];
#pragma unroll
            for (int j = 0; j < 4; ++j) {
                FragAB b;
                const int n = j * 16 + ln;
                b.q[0] = *(const uint4*)&Btb[n * 64 + c * 32 + 8 * half];
                b.q[1] = *(const uint4*)&Btb[n * 64 + c * 32 + 16 + 8 * half];
                acc[j] = __builtin_amdgcn_wmma_f32_16x16x32_bf16(
                    false, a.v, false, b.v, (short)0, acc[j], false, false);
            }
        }
        __syncthreads();
    }

    if (mode == 0) {
#pragma unroll
        for (int j = 0; j < 4; ++j)
#pragma unroll
            for (int r = 0; r < 8; ++r) {
                int row = bm + wave * 16 + r + 8 * half;
                int col = bn + j * 16 + ln;
                ((unsigned short*)out)[(size_t)row * DMODEL + col] = f2bf(acc[j][r]);
            }
    } else {
#pragma unroll
        for (int j = 0; j < 4; ++j)
#pragma unroll
            for (int r = 0; r < 8; ++r) {
                int row = bm + wave * 16 + r + 8 * half;
                int col = bn + j * 16 + ln;
                ((float*)out)[(size_t)row * DMODEL + col] = acc[j][r] + bias[col];
            }
    }
}

// ---------------------------------------------------------------------------
// Flash attention per (segment, head). Block = 4 waves = 64 query rows.
// Q/K/V are bf16 [T][512] in workspace; O is f32 [T][512].
// K chunk staged with async global->LDS b128 copies (ASYNCcnt).
// ---------------------------------------------------------------------------
__global__ void __launch_bounds__(128)
attn_wmma(const unsigned short* __restrict__ Q,
          const unsigned short* __restrict__ K,
          const unsigned short* __restrict__ V,
          float* __restrict__ O,
          const int* __restrict__ offsets)
{
    __shared__ alignas(16) unsigned short Kbuf[32 * 64];      // [token][dh]
    __shared__ alignas(16) unsigned short Vtb[64 * 32];       // [dh][token]
    __shared__ alignas(16) unsigned short Pbuf[4 * 16 * 32];  // per-wave P tile

    const int b   = blockIdx.y;
    const int h   = blockIdx.z;
    const int off = offsets[b];
    const int len = offsets[b + 1] - off;
    const int q0  = blockIdx.x * 64;
    if (q0 >= len) return;

    const int tid  = threadIdx.x;
    const int lane = tid & 31;
    const int wave = tid >> 5;
    const int half = lane >> 4;
    const int ln   = lane & 15;

    // Q A-fragments (row = lane%16), dh chunks c=0,1 -- kept in registers all loop
    const int qlocal   = min(q0 + wave * 16 + ln, len - 1);
    const size_t qbase = (size_t)(off + qlocal) * DMODEL + h * DH;
    FragAB qf[2];
#pragma unroll
    for (int c = 0; c < 2; ++c) {
        qf[c].q[0] = *(const uint4*)&Q[qbase + c * 32 + 8 * half];
        qf[c].q[1] = *(const uint4*)&Q[qbase + c * 32 + 16 + 8 * half];
    }

    float m[8], l[8];
#pragma unroll
    for (int r = 0; r < 8; ++r) { m[r] = -1e30f; l[r] = 0.0f; }
    f32x8 oacc[4];
#pragma unroll
    for (int j = 0; j < 4; ++j)
#pragma unroll
        for (int r = 0; r < 8; ++r) oacc[j][r] = 0.0f;

    for (int kk = 0; kk < len; kk += 32) {
        // stage K chunk (32 tokens x 64 dh) row-major via async b128 copies
#pragma unroll
        for (int i = 0; i < 2; ++i) {
            int idx = tid + i * 128;
            int row = idx >> 3;
            int c4  = idx & 7;
            int tg  = off + min(kk + row, len - 1);
            async_copy_b128(&Kbuf[row * 64 + c4 * 8],
                            &K[(size_t)tg * DMODEL + h * DH + c4 * 8]);
        }
        // stage V chunk transposed: Vtb[dh][token]
#pragma unroll
        for (int i = 0; i < 16; ++i) {
            int idx = tid + i * 128;
            int row = idx >> 6;   // token
            int n   = idx & 63;   // dh
            int tg  = off + min(kk + row, len - 1);
            Vtb[n * 32 + row] = V[(size_t)tg * DMODEL + h * DH + n];
        }
        wait_asynccnt0();
        __syncthreads();

        // S = Q * K^T : two 16x16 tiles, K-dim = dh = 64 -> two chained WMMAs
        f32x8 s[2];
#pragma unroll
        for (int t = 0; t < 2; ++t) {
#pragma unroll
            for (int r = 0; r < 8; ++r) s[t][r] = 0.0f;
#pragma unroll
            for (int c = 0; c < 2; ++c) {
                FragAB kf;
                const int krow = t * 16 + ln;  // B: N = key token (lane%16)
                kf.q[0] = *(const uint4*)&Kbuf[krow * 64 + c * 32 + 8 * half];
                kf.q[1] = *(const uint4*)&Kbuf[krow * 64 + c * 32 + 16 + 8 * half];
                s[t] = __builtin_amdgcn_wmma_f32_16x16x32_bf16(
                    false, qf[c].v, false, kf.v, (short)0, s[t], false, false);
            }
        }

        // online softmax over this 32-key chunk (rows live in C layout)
        const bool v0 = (kk + ln) < len;
        const bool v1 = (kk + 16 + ln) < len;
#pragma unroll
        for (int r = 0; r < 8; ++r) {
            float s0 = v0 ? s[0][r] * 0.125f : -1e30f;   // scale = dh^-0.5 = 1/8
            float s1 = v1 ? s[1][r] * 0.125f : -1e30f;
            float rm = fmaxf(s0, s1);
#pragma unroll
            for (int d = 1; d < 16; d <<= 1) rm = fmaxf(rm, __shfl_xor(rm, d, 32));
            float mn    = fmaxf(m[r], rm);
            float alpha = __expf(m[r] - mn);
            m[r] = mn;
            float p0 = __expf(s0 - mn);
            float p1 = __expf(s1 - mn);
            float rs = p0 + p1;
#pragma unroll
            for (int d = 1; d < 16; d <<= 1) rs += __shfl_xor(rs, d, 32);
            l[r] = l[r] * alpha + rs;
#pragma unroll
            for (int j = 0; j < 4; ++j) oacc[j][r] *= alpha;
            // park P in LDS in row-major [16][32] for C->A relayout
            Pbuf[wave * 512 + (r + 8 * half) * 32 + ln]      = f2bf(p0);
            Pbuf[wave * 512 + (r + 8 * half) * 32 + 16 + ln] = f2bf(p1);
        }
        __syncthreads();

        // O += P * V : A = P (16x32), B = V (32x16 per dh tile)
        FragAB pf;
        pf.q[0] = *(const uint4*)&Pbuf[wave * 512 + ln * 32 + 8 * half];
        pf.q[1] = *(const uint4*)&Pbuf[wave * 512 + ln * 32 + 16 + 8 * half];
#pragma unroll
        for (int j = 0; j < 4; ++j) {
            FragAB vf;
            const int n = j * 16 + ln;
            vf.q[0] = *(const uint4*)&Vtb[n * 32 + 8 * half];
            vf.q[1] = *(const uint4*)&Vtb[n * 32 + 16 + 8 * half];
            oacc[j] = __builtin_amdgcn_wmma_f32_16x16x32_bf16(
                false, pf.v, false, vf.v, (short)0, oacc[j], false, false);
        }
        __syncthreads();
    }

    // write O (f32) for valid query rows
#pragma unroll
    for (int j = 0; j < 4; ++j) {
#pragma unroll
        for (int r = 0; r < 8; ++r) {
            int qrow = q0 + wave * 16 + r + 8 * half;
            if (qrow < len) {
                O[(size_t)(off + qrow) * DMODEL + h * DH + j * 16 + ln] =
                    oacc[j][r] * (1.0f / l[r]);
            }
        }
    }
}

// ---------------------------------------------------------------------------
extern "C" void kernel_launch(void* const* d_in, const int* in_sizes, int n_in,
                              void* d_out, int out_size, void* d_ws, size_t ws_size,
                              hipStream_t stream)
{
    const float* x  = (const float*)d_in[0];
    const float* Wq = (const float*)d_in[1];
    const float* Wk = (const float*)d_in[2];
    const float* Wv = (const float*)d_in[3];
    const float* Wo = (const float*)d_in[4];
    const float* bo = (const float*)d_in[5];
    const int* offs = (const int*)d_in[6];

    const int T = in_sizes[0] / DMODEL;   // 9728 (multiple of 128)
    const int B = in_sizes[6] - 1;        // 8 segments

    // workspace: Q,K,V bf16 [T][512] then O f32 [T][512]  (~50 MB)
    unsigned short* Qw = (unsigned short*)d_ws;
    unsigned short* Kw = Qw + (size_t)T * DMODEL;
    unsigned short* Vw = Kw + (size_t)T * DMODEL;
    float*          Ow = (float*)(Vw + (size_t)T * DMODEL);

    dim3 gGemm(T / 128, DMODEL / 64);
    gemm512_wmma<<<gGemm, 256, 0, stream>>>(x, Wq, nullptr, Qw, 0);
    gemm512_wmma<<<gGemm, 256, 0, stream>>>(x, Wk, nullptr, Kw, 0);
    gemm512_wmma<<<gGemm, 256, 0, stream>>>(x, Wv, nullptr, Vw, 0);

    dim3 gAttn((SEG_LMAX + 63) / 64, B, HEADS);
    attn_wmma<<<gAttn, 128, 0, stream>>>(Qw, Kw, Vw, Ow, offs);

    gemm512_wmma<<<gGemm, 256, 0, stream>>>(Ow, Wo, bo, d_out, 1);
}
